// OptimizedMultiHeadAttn_18545668784947
// MI455X (gfx1250) — matrix operand order repaired and checked by reading, not compiled, not run on hardware
//
#include <hip/hip_runtime.h>
#include <hip/hip_bf16.h>

// ---------------------------------------------------------------------------
// Types
// ---------------------------------------------------------------------------
typedef __bf16 bf16_t;
typedef __attribute__((ext_vector_type(16))) __bf16 v16bf;
typedef __attribute__((ext_vector_type(8)))  __bf16 v8bf;
typedef __attribute__((ext_vector_type(8)))  float  v8f;

#define WMMA_BF16(A, B, C) \
  __builtin_amdgcn_wmma_f32_16x16x32_bf16(false, (A), false, (B), (short)0, (C), false, false)

// async global->LDS copy of 16 bytes (GVS mode: saddr base + 32-bit voffset)
#define ASYNC_B128(ldsoff, voff, base)                                        \
  asm volatile("global_load_async_to_lds_b128 %0, %1, %2"                     \
               :: "v"(ldsoff), "v"(voff), "s"(base) : "memory")

#define WAIT_ASYNC0() asm volatile("s_wait_asynccnt 0x0" ::: "memory")

static constexpr int kB  = 2;
static constexpr int kS  = 2048;
static constexpr int kD  = 1024;
static constexpr int kH  = 16;
static constexpr int kDK = 64;
static constexpr int kHD = kH * kDK;   // 1024
static constexpr int kQKV = 3 * kHD;   // 3072
static constexpr int kM  = kB * kS;    // 4096

// GEMM block tiling: 8 waves (256 thr) -> 128(M) x 128(N), K-step 32
static constexpr int TM  = 128;
static constexpr int TN  = 128;
static constexpr int TK  = 32;
static constexpr int LDA = 40;   // padded LDS row stride (80 B, 16B-aligned, conflict-free)

// Attention LDS strides
static constexpr int LDK = 72;   // K tile rows (64 dk elems + pad), 144 B
static constexpr int LDV = 40;   // V tile rows (32 s elems + pad), 80 B
static constexpr int LDP = 40;   // P tile rows (32 key elems + pad), 80 B

// ---------------------------------------------------------------------------
// Helpers
// ---------------------------------------------------------------------------
static __device__ __forceinline__ bf16_t f2bf(float f) {
  unsigned u = __builtin_bit_cast(unsigned, f);
  u += 0x7FFFu + ((u >> 16) & 1u);               // round-to-nearest-even
  unsigned short h = (unsigned short)(u >> 16);
  return __builtin_bit_cast(bf16_t, h);
}

static __device__ __forceinline__ v8f zero_v8f() {
  v8f z;
#pragma unroll
  for (int i = 0; i < 8; ++i) z[i] = 0.0f;
  return z;
}

// A-fragment (16x32 bf16): lane -> M row; K pattern {klo..klo+7, 16+klo..16+klo+7}
static __device__ __forceinline__ v16bf load_frag_a(const bf16_t* base, int stride) {
  const int lane = threadIdx.x & 31;
  const int m    = lane & 15;
  const int klo  = (lane >> 4) << 3;
  const bf16_t* p = base + (size_t)m * stride + klo;
  v8bf lo = *(const v8bf*)(p);
  v8bf hi = *(const v8bf*)(p + 16);
  v16bf r;
#pragma unroll
  for (int i = 0; i < 8; ++i) { r[i] = lo[i]; r[i + 8] = hi[i]; }
  return r;
}

// B-fragment (32x16 bf16) from an [N, K] row-major tile: lane -> N column;
// K pattern {klo..klo+7, 16+klo..16+klo+7}, klo = 8*(lane/16).
static __device__ __forceinline__ v16bf load_frag_b(const bf16_t* base, int stride) {
  const int lane = threadIdx.x & 31;
  const int n    = lane & 15;
  const int k0   = (lane >> 4) << 3;
  const bf16_t* p = base + (size_t)n * stride + k0;
  v8bf lo = *(const v8bf*)(p);
  v8bf hi = *(const v8bf*)(p + 16);
  v16bf r;
#pragma unroll
  for (int i = 0; i < 8; ++i) { r[i] = lo[i]; r[i + 8] = hi[i]; }
  return r;
}

// ---------------------------------------------------------------------------
// fp32 -> bf16 conversion
// ---------------------------------------------------------------------------
__global__ void __launch_bounds__(256)
cvt_f32_bf16_kernel(const float* __restrict__ in, bf16_t* __restrict__ out, int n) {
  int i = (blockIdx.x * blockDim.x + threadIdx.x) * 4;
  if (i + 3 < n) {
    float4 v = *(const float4*)(in + i);
    out[i + 0] = f2bf(v.x);
    out[i + 1] = f2bf(v.y);
    out[i + 2] = f2bf(v.z);
    out[i + 3] = f2bf(v.w);
  } else {
    for (; i < n; ++i) out[i] = f2bf(in[i]);
  }
}

// ---------------------------------------------------------------------------
// Multi-wave GEMM core: 256 threads compute a 128x128 block of A[M,K] @ W[N,K]^T.
// A/W K-step tiles staged to LDS with async copies, double buffered.
// ---------------------------------------------------------------------------
static __device__ __forceinline__ void gemm_tile_core(const bf16_t* __restrict__ Ab,
                                                      const bf16_t* __restrict__ Wb,
                                                      int K,
                                                      bf16_t* shA, bf16_t* shW,
                                                      v8f acc[2][4]) {
  const int tid = threadIdx.x;
  const int w   = tid >> 5;
  const int wm  = w >> 1;          // 0..3
  const int wn  = w & 1;           // 0..1

#pragma unroll
  for (int mi = 0; mi < 2; ++mi)
#pragma unroll
    for (int nf = 0; nf < 4; ++nf) acc[mi][nf] = zero_v8f();

  // Per-thread copy chunks: 512 x 16B per tile, 2 chunks/thread.
  const int i0 = tid, i1 = tid + 256;
  const int r0 = i0 >> 2, q0 = i0 & 3;
  const int r1 = i1 >> 2, q1 = i1 & 3;
  const unsigned g0 = (unsigned)((r0 * K + q0 * 8) * 2);
  const unsigned g1 = (unsigned)((r1 * K + q1 * 8) * 2);
  const unsigned l0 = (unsigned)((r0 * LDA + q0 * 8) * 2);
  const unsigned l1 = (unsigned)((r1 * LDA + q1 * 8) * 2);

  const unsigned ldsA = (unsigned)(size_t)shA;             // low 32 bits == LDS offset
  const unsigned ldsW = (unsigned)(size_t)shW;
  const unsigned bufA = (unsigned)(TM * LDA * 2);
  const unsigned bufW = (unsigned)(TN * LDA * 2);

  ASYNC_B128(ldsA + l0, g0, Ab);
  ASYNC_B128(ldsA + l1, g1, Ab);
  ASYNC_B128(ldsW + l0, g0, Wb);
  ASYNC_B128(ldsW + l1, g1, Wb);

  const int NT = K / TK;
  for (int t = 0; t < NT; ++t) {
    const int cur = t & 1;
    WAIT_ASYNC0();
    __syncthreads();

    if (t + 1 < NT) {
      const int nxt = cur ^ 1;
      const unsigned kb = (unsigned)((t + 1) * TK * 2);
      ASYNC_B128(ldsA + nxt * bufA + l0, g0 + kb, Ab);
      ASYNC_B128(ldsA + nxt * bufA + l1, g1 + kb, Ab);
      ASYNC_B128(ldsW + nxt * bufW + l0, g0 + kb, Wb);
      ASYNC_B128(ldsW + nxt * bufW + l1, g1 + kb, Wb);
    }

    const bf16_t* At = shA + cur * (TM * LDA) + (wm * 32) * LDA;
    const bf16_t* Wt = shW + cur * (TN * LDA) + (wn * 64) * LDA;
    v16bf a0 = load_frag_a(At, LDA);
    v16bf a1 = load_frag_a(At + 16 * LDA, LDA);
#pragma unroll
    for (int nf = 0; nf < 4; ++nf) {
      v16bf bb = load_frag_b(Wt + nf * 16 * LDA, LDA);
      acc[0][nf] = WMMA_BF16(a0, bb, acc[0][nf]);
      acc[1][nf] = WMMA_BF16(a1, bb, acc[1][nf]);
    }
  }
}

// ---------------------------------------------------------------------------
// QKV projection: qkv = x @ attn_w^T + attn_b, scattered into
//   Q  [B,H,S,DK] bf16 (pre-scaled by 1/sqrt(DK)),  K [B,H,S,DK] bf16,
//   Vt [B,H,DK,S] bf16 (transposed so P@V B-fragments are contiguous)
// ---------------------------------------------------------------------------
__global__ void __launch_bounds__(256)
gemm_qkv_kernel(const bf16_t* __restrict__ A, const bf16_t* __restrict__ W,
                const float* __restrict__ bias,
                bf16_t* __restrict__ Qo, bf16_t* __restrict__ Ko, bf16_t* __restrict__ Vto) {
  __shared__ __align__(16) bf16_t shA[2 * TM * LDA];
  __shared__ __align__(16) bf16_t shW[2 * TN * LDA];

  const int lane = threadIdx.x & 31;
  const int half = lane >> 4;
  const int nlan = lane & 15;
  const int w  = threadIdx.x >> 5;
  const int wm = w >> 1, wn = w & 1;
  const int n0 = blockIdx.x * TN;
  const int m0 = blockIdx.y * TM;

  v8f acc[2][4];
  gemm_tile_core(A + (size_t)m0 * kD, W + (size_t)n0 * kD, kD, shA, shW, acc);

#pragma unroll
  for (int nf = 0; nf < 4; ++nf) {
    const int n   = n0 + wn * 64 + nf * 16 + nlan;
    const float bn = bias[n];
    const int sec = n >> 10;       // 0 = q, 1 = k, 2 = v
    const int nn  = n & 1023;
    const int h   = nn >> 6;
    const int dk  = nn & 63;
#pragma unroll
    for (int mi = 0; mi < 2; ++mi) {
#pragma unroll
      for (int r = 0; r < 8; ++r) {
        const int m = m0 + wm * 32 + mi * 16 + r + 8 * half;  // global row = b*S + s
        const int b = m >> 11;
        const int s = m & (kS - 1);
        const float v = acc[mi][nf][r] + bn;
        const size_t bh = (size_t)(b * kH + h);
        if (sec == 0)      Qo[(bh * kS + s) * kDK + dk]  = f2bf(v * 0.125f);
        else if (sec == 1) Ko[(bh * kS + s) * kDK + dk]  = f2bf(v);
        else               Vto[(bh * kDK + dk) * kS + s] = f2bf(v);
      }
    }
  }
}

// ---------------------------------------------------------------------------
// Output projection: out = attn_out @ proj_w^T + proj_b  (fp32 result)
// ---------------------------------------------------------------------------
__global__ void __launch_bounds__(256)
gemm_proj_kernel(const bf16_t* __restrict__ A, const bf16_t* __restrict__ W,
                 const float* __restrict__ bias, float* __restrict__ out) {
  __shared__ __align__(16) bf16_t shA[2 * TM * LDA];
  __shared__ __align__(16) bf16_t shW[2 * TN * LDA];

  const int lane = threadIdx.x & 31;
  const int half = lane >> 4;
  const int nlan = lane & 15;
  const int w  = threadIdx.x >> 5;
  const int wm = w >> 1, wn = w & 1;
  const int n0 = blockIdx.x * TN;
  const int m0 = blockIdx.y * TM;

  v8f acc[2][4];
  gemm_tile_core(A + (size_t)m0 * kHD, W + (size_t)n0 * kHD, kHD, shA, shW, acc);

#pragma unroll
  for (int nf = 0; nf < 4; ++nf) {
    const int n = n0 + wn * 64 + nf * 16 + nlan;
    const float bn = bias[n];
#pragma unroll
    for (int mi = 0; mi < 2; ++mi) {
#pragma unroll
      for (int r = 0; r < 8; ++r) {
        const int m = m0 + wm * 32 + mi * 16 + r + 8 * half;
        out[(size_t)m * kD + n] = acc[mi][nf][r] + bn;
      }
    }
  }
}

// ---------------------------------------------------------------------------
// Causal flash attention, 4 waves/block sharing async-staged K/V tiles.
// Wave w handles q-tile iq = blockIdx.x*4 + w (16 rows). Per kv-step (32 keys):
// stage K(32x64) and V(64x32) into LDS (double buffered), scores (4 WMMA),
// online softmax, P via per-wave LDS slice, P@V (4 WMMA).
// ---------------------------------------------------------------------------
__global__ void __launch_bounds__(128)
attn_fwd_kernel(const bf16_t* __restrict__ Q, const bf16_t* __restrict__ Km,
                const bf16_t* __restrict__ Vt, bf16_t* __restrict__ Obf) {
  __shared__ __align__(16) bf16_t shK[2 * 32 * LDK];   // 9216 B
  __shared__ __align__(16) bf16_t shV[2 * 64 * LDV];   // 10240 B
  __shared__ __align__(16) bf16_t shP[4 * 16 * LDP];   // 5120 B

  const int tid  = threadIdx.x;
  const int wv   = tid >> 5;           // wave 0..3
  const int lane = tid & 31;
  const int half = lane >> 4;
  const int nlan = lane & 15;
  const int iq = blockIdx.x * 4 + wv;  // this wave's q tile (16 rows)
  const int bh = blockIdx.y;           // b*H + h
  const int b  = bh >> 4;
  const int h  = bh & 15;

  const bf16_t* Kb = Km + (size_t)bh * kS * kDK;   // [s][dk]
  const bf16_t* Vb = Vt + (size_t)bh * kDK * kS;   // [dk][s]

  // --- staging chunk geometry (128 threads, 2 chunks each per tile) ---
  // K tile: 32 rows x 64 dk -> 256 x 16B chunks
  const int ik1 = tid + 128;
  const int kr0 = tid >> 3, kq0 = tid & 7;
  const int kr1 = ik1 >> 3, kq1 = ik1 & 7;
  const unsigned gK0 = (unsigned)((kr0 * kDK + kq0 * 8) * 2);
  const unsigned gK1 = (unsigned)((kr1 * kDK + kq1 * 8) * 2);
  const unsigned lK0 = (unsigned)((kr0 * LDK + kq0 * 8) * 2);
  const unsigned lK1 = (unsigned)((kr1 * LDK + kq1 * 8) * 2);
  // V tile: 64 rows x 32 s -> 256 x 16B chunks
  const int vr0 = tid >> 2, vq0 = tid & 3;
  const int vr1 = ik1 >> 2, vq1 = ik1 & 3;
  const unsigned gV0 = (unsigned)((vr0 * kS + vq0 * 8) * 2);
  const unsigned gV1 = (unsigned)((vr1 * kS + vq1 * 8) * 2);
  const unsigned lV0 = (unsigned)((vr0 * LDV + vq0 * 8) * 2);
  const unsigned lV1 = (unsigned)((vr1 * LDV + vq1 * 8) * 2);

  const unsigned ldsK = (unsigned)(size_t)shK;
  const unsigned ldsV = (unsigned)(size_t)shV;
  const unsigned bufK = (unsigned)(32 * LDK * 2);
  const unsigned bufV = (unsigned)(64 * LDV * 2);

  auto stage = [&](int j, int buf) {
    const unsigned jK = (unsigned)(j * 32 * kDK * 2);   // j*4096 B
    const unsigned jV = (unsigned)(j * 32 * 2);         // j*64 B
    ASYNC_B128(ldsK + buf * bufK + lK0, gK0 + jK, Kb);
    ASYNC_B128(ldsK + buf * bufK + lK1, gK1 + jK, Kb);
    ASYNC_B128(ldsV + buf * bufV + lV0, gV0 + jV, Vb);
    ASYNC_B128(ldsV + buf * bufV + lV1, gV1 + jV, Vb);
  };

  const bf16_t* Qb = Q + ((size_t)bh * kS + iq * 16) * kDK;
  const v16bf qa0 = load_frag_a(Qb + 0,  kDK);   // dk 0..31
  const v16bf qa1 = load_frag_a(Qb + 32, kDK);   // dk 32..63

  v8f o0 = zero_v8f(), o1 = zero_v8f(), o2 = zero_v8f(), o3 = zero_v8f();
  float mst[8], lst[8];
#pragma unroll
  for (int r = 0; r < 8; ++r) { mst[r] = -1e30f; lst[r] = 0.0f; }

  bf16_t* pSh = shP + wv * (16 * LDP);           // per-wave private P slice

  const int myJmax  = iq >> 1;
  const int blkJmax = (blockIdx.x * 4 + 3) >> 1; // jmax of the last wave in block

  stage(0, 0);
  for (int j = 0; j <= blkJmax; ++j) {
    const int cur = j & 1;
    WAIT_ASYNC0();        // my async copies for buffer `cur` are done
    __syncthreads();      // all copies visible; all reads of `cur^1` finished

    if (j + 1 <= blkJmax) stage(j + 1, cur ^ 1);

    if (j <= myJmax) {
      const bf16_t* Kt = shK + cur * (32 * LDK);
      v8f s0 = zero_v8f(), s1 = zero_v8f();
      {
        v16bf bk0 = load_frag_b(Kt + 0,             LDK);  // keys 0-15, dk 0-31
        v16bf bk1 = load_frag_b(Kt + 32,            LDK);  // keys 0-15, dk 32-63
        s0 = WMMA_BF16(qa0, bk0, s0);
        s0 = WMMA_BF16(qa1, bk1, s0);
        v16bf bk2 = load_frag_b(Kt + 16 * LDK + 0,  LDK);  // keys 16-31
        v16bf bk3 = load_frag_b(Kt + 16 * LDK + 32, LDK);
        s1 = WMMA_BF16(qa0, bk2, s1);
        s1 = WMMA_BF16(qa1, bk3, s1);
      }

      const int colA = j * 32 + nlan;
      const int colB = colA + 16;
#pragma unroll
      for (int r = 0; r < 8; ++r) {
        const int row = iq * 16 + r + 8 * half;
        float e0 = (colA <= row) ? s0[r] : -1e30f;   // causal mask
        float e1 = (colB <= row) ? s1[r] : -1e30f;

        float mx = fmaxf(e0, e1);
#pragma unroll
        for (int d = 8; d >= 1; d >>= 1) mx = fmaxf(mx, __shfl_xor(mx, d, 32));

        const float mnew = fmaxf(mst[r], mx);
        const float corr = __expf(mst[r] - mnew);
        const float p0 = __expf(e0 - mnew);
        const float p1 = __expf(e1 - mnew);
        float rs = p0 + p1;
#pragma unroll
        for (int d = 8; d >= 1; d >>= 1) rs += __shfl_xor(rs, d, 32);

        lst[r] = lst[r] * corr + rs;
        mst[r] = mnew;
        o0[r] *= corr; o1[r] *= corr; o2[r] *= corr; o3[r] *= corr;

        pSh[(r + 8 * half) * LDP + nlan]      = f2bf(p0);
        pSh[(r + 8 * half) * LDP + nlan + 16] = f2bf(p1);
      }

      // P (16x32) re-read in A-fragment layout; per-wave slice, DS ops are
      // in-order within a wave -> no barrier needed.
      v16bf pa;
      {
        const int klo = half << 3;
        const bf16_t* pp = &pSh[nlan * LDP + klo];
        v8bf lo = *(const v8bf*)(pp);
        v8bf hi = *(const v8bf*)(pp + 16);
#pragma unroll
        for (int i2 = 0; i2 < 8; ++i2) { pa[i2] = lo[i2]; pa[i2 + 8] = hi[i2]; }
      }

      const bf16_t* Vtile = shV + cur * (64 * LDV);   // rows = dk, cols = key
      v16bf bv0 = load_frag_b(Vtile + 0  * 16 * LDV, LDV);
      v16bf bv1 = load_frag_b(Vtile + 1  * 16 * LDV, LDV);
      v16bf bv2 = load_frag_b(Vtile + 2  * 16 * LDV, LDV);
      v16bf bv3 = load_frag_b(Vtile + 3  * 16 * LDV, LDV);
      o0 = WMMA_BF16(pa, bv0, o0);
      o1 = WMMA_BF16(pa, bv1, o1);
      o2 = WMMA_BF16(pa, bv2, o2);
      o3 = WMMA_BF16(pa, bv3, o3);
    }
  }

  // normalize and store as bf16 [B, S, H*DK] for the projection GEMM
#pragma unroll
  for (int r = 0; r < 8; ++r) {
    const float inv = 1.0f / lst[r];
    const int srow = iq * 16 + r + 8 * half;
    const size_t base = ((size_t)b * kS + srow) * kHD + h * kDK + nlan;
    Obf[base + 0]  = f2bf(o0[r] * inv);
    Obf[base + 16] = f2bf(o1[r] * inv);
    Obf[base + 32] = f2bf(o2[r] * inv);
    Obf[base + 48] = f2bf(o3[r] * inv);
  }
}

// ---------------------------------------------------------------------------
// Launch
// ---------------------------------------------------------------------------
extern "C" void kernel_launch(void* const* d_in, const int* in_sizes, int n_in,
                              void* d_out, int out_size, void* d_ws, size_t ws_size,
                              hipStream_t stream) {
  (void)in_sizes; (void)n_in; (void)out_size; (void)ws_size;

  const float* x      = (const float*)d_in[0];
  const float* attn_w = (const float*)d_in[1];
  const float* attn_b = (const float*)d_in[2];
  const float* proj_w = (const float*)d_in[3];
  const float* proj_b = (const float*)d_in[4];
  float* out = (float*)d_out;

  // Workspace layout (all bf16, ~48 MB total)
  bf16_t* p    = (bf16_t*)d_ws;
  bf16_t* xb   = p; p += (size_t)kM * kD;            // 4096x1024
  bf16_t* wqkv = p; p += (size_t)kQKV * kD;          // 3072x1024
  bf16_t* wprj = p; p += (size_t)kD * kHD;           // 1024x1024
  bf16_t* qb   = p; p += (size_t)kB * kH * kS * kDK; // Q  [B,H,S,DK]
  bf16_t* kb   = p; p += (size_t)kB * kH * kS * kDK; // K  [B,H,S,DK]
  bf16_t* vtb  = p; p += (size_t)kB * kH * kS * kDK; // Vt [B,H,DK,S]
  bf16_t* aob  = p; p += (size_t)kM * kHD;           // attn out [B*S, H*DK]

  const int nx = kM * kD, nw = kQKV * kD, npw = kD * kHD;
  cvt_f32_bf16_kernel<<<dim3((nx / 4 + 255) / 256), dim3(256), 0, stream>>>(x, xb, nx);
  cvt_f32_bf16_kernel<<<dim3((nw / 4 + 255) / 256), dim3(256), 0, stream>>>(attn_w, wqkv, nw);
  cvt_f32_bf16_kernel<<<dim3((npw / 4 + 255) / 256), dim3(256), 0, stream>>>(proj_w, wprj, npw);

  gemm_qkv_kernel<<<dim3(kQKV / TN, kM / TM), dim3(256), 0, stream>>>(xb, wqkv, attn_b, qb, kb, vtb);
  attn_fwd_kernel<<<dim3(kS / 64, kB * kH), dim3(128), 0, stream>>>(qb, kb, vtb, aob);
  gemm_proj_kernel<<<dim3(kD / TN, kM / TM), dim3(256), 0, stream>>>(aob, wprj, proj_b, out);
}
